// LSTMModel_13176959664668
// MI455X (gfx1250) — compile-verified, matrix operand order checked by reading, not run on hardware
//
#include <hip/hip_runtime.h>
#include <hip/hip_bf16.h>
#include <math.h>

typedef _Float16 v16h __attribute__((ext_vector_type(16)));
typedef _Float16 v8h  __attribute__((ext_vector_type(8)));
typedef float    v8f  __attribute__((ext_vector_type(8)));

// LDS layout (bytes), total 125952:
//   wb   @ 0       : 112 B-fragments * 512 f16 (1KB each)  = 114688
//   hab  @ 114688  : layer-0 h, 16 rows x 72 f16 (padded)  = 2304
//   hbb  @ 116992  : layer-1 h, 16 rows x 72 f16           = 2304
//   xb   @ 119296  : 2 x (16 rows x 40 f16) x_t staging    = 2560
//   hfin @ 121856  : final h2 in f32, 16 x 64               = 4096
#define SMEM_BYTES 125952

__device__ __forceinline__ v16h ld_frag(const _Float16* p) {
  v8h lo = *(const v8h*)(p);
  v8h hi = *(const v8h*)(p + 8);
  return __builtin_shufflevector(lo, hi, 0,1,2,3,4,5,6,7,8,9,10,11,12,13,14,15);
}

// Branch-free transcendentals: native v_tanh_f32 on gfx1250 if the builtin
// exists, otherwise v_exp_f32 (exp2) + v_rcp_f32. No libm, no exec juggling.
#if defined(__has_builtin) && __has_builtin(__builtin_amdgcn_tanhf)
__device__ __forceinline__ float fast_tanh(float xv) {
  return __builtin_amdgcn_tanhf(xv);
}
#else
__device__ __forceinline__ float fast_tanh(float xv) {
  float e = __builtin_amdgcn_exp2f(xv * -2.8853900817779268f);  // exp(-2x)
  return (1.0f - e) * __builtin_amdgcn_rcpf(1.0f + e);
}
#endif

__device__ __forceinline__ float fast_sig(float xv) {
  float e = __builtin_amdgcn_exp2f(xv * -1.4426950408889634f);  // exp(-x)
  return __builtin_amdgcn_rcpf(1.0f + e);
}

__device__ __forceinline__ v8f wmma_f16(v16h a, v16h b, v8f c) {
  return __builtin_amdgcn_wmma_f32_16x16x32_f16(false, a, false, b, (short)0, c,
                                                false, false);
}

extern "C" __global__ void __launch_bounds__(128)
lstm2_fused(const float* __restrict__ x,
            const float* __restrict__ Wih0, const float* __restrict__ Whh0,
            const float* __restrict__ bih0, const float* __restrict__ bhh0,
            const float* __restrict__ Wih1, const float* __restrict__ Whh1,
            const float* __restrict__ bih1, const float* __restrict__ bhh1,
            const float* __restrict__ fcw,  const float* __restrict__ fcb,
            float* __restrict__ out)
{
  extern __shared__ char smem[];
  _Float16* wb   = (_Float16*)(smem);
  _Float16* hab  = (_Float16*)(smem + 114688);
  _Float16* hbb  = (_Float16*)(smem + 116992);
  _Float16* xb   = (_Float16*)(smem + 119296);
  float*    hfin = (float*)   (smem + 121856);

  const int tid  = threadIdx.x;
  const int wave = tid >> 5;
  const int lane = tid & 31;
  const int hl   = lane >> 4;      // K half within fragment
  const int m    = lane & 15;      // row (A) / column (B,C) index
  const int b0   = blockIdx.x * 16;

  // ---- pack weights into per-lane B-fragment layout (f32 -> f16) ----
  // frag ids: [0,16) Wih0 (K padded 23->32); [16,48) Whh0; [48,80) Wih1; [80,112) Whh1
  for (int f = wave; f < 112; f += 4) {
    const float* src; int ldw, Ks, nt, kc;
    if (f < 16)      { src = Wih0; ldw = 23; Ks = 23; nt = f;           kc = 0;          }
    else if (f < 48) { src = Whh0; ldw = 64; Ks = 64; nt = (f-16) >> 1; kc = (f-16) & 1; }
    else if (f < 80) { src = Wih1; ldw = 64; Ks = 64; nt = (f-48) >> 1; kc = (f-48) & 1; }
    else             { src = Whh1; ldw = 64; Ks = 64; nt = (f-80) >> 1; kc = (f-80) & 1; }
    const int row   = nt * 16 + m;        // gate-output unit 0..255
    const int kbase = kc * 32 + hl * 16;
    v8h lo, hi;
#pragma unroll
    for (int k = 0; k < 8; ++k) {
      int k0 = kbase + k, k1 = kbase + 8 + k;
      lo[k] = (k0 < Ks) ? (_Float16)src[row * ldw + k0] : (_Float16)0.0f;
      hi[k] = (k1 < Ks) ? (_Float16)src[row * ldw + k1] : (_Float16)0.0f;
    }
    _Float16* dst = wb + f * 512 + lane * 16;
    *(v8h*)dst       = lo;
    *(v8h*)(dst + 8) = hi;
  }

  // ---- zero h state and x staging (pad columns must stay zero) ----
  for (int i = tid; i < 1152; i += 128) { hab[i] = (_Float16)0.0f; hbb[i] = (_Float16)0.0f; }
  for (int i = tid; i < 1280; i += 128) xb[i] = (_Float16)0.0f;

  // ---- per-wave gate biases (column n = wave*16 + m of each gate block) ----
  float bias0[4], bias1[4];
#pragma unroll
  for (int g = 0; g < 4; ++g) {
    int col = g * 64 + wave * 16 + m;
    bias0[g] = bih0[col] + bhh0[col];
    bias1[g] = bih1[col] + bhh1[col];
  }

  // ---- x gather plan: 368 floats/step (16 rows x 23), 128 threads x 3 ----
  // Fully branch-free: invalid third element loads x[0] and stores to a dead
  // pad slot (column 39 is never read by the A-fragment loads).
  const int e0 = tid, e1 = tid + 128, e2 = tid + 256;
  const int r0 = e0 / 23, q0 = e0 % 23;
  const int r1 = e1 / 23, q1 = e1 % 23;
  const int r2 = e2 / 23, q2 = e2 % 23;
  const bool v2 = (e2 < 368);
  const float* p0 = x + ((size_t)(b0 + r0) * 512) * 23 + q0;
  const float* p1 = x + ((size_t)(b0 + r1) * 512) * 23 + q1;
  const float* p2 = v2 ? (x + ((size_t)(b0 + r2) * 512) * 23 + q2) : x;
  const int d0 = r0 * 40 + q0;
  const int d1 = r1 * 40 + q1;
  const int d2 = v2 ? (r2 * 40 + q2) : 39;   // dead pad slot when invalid

  __syncthreads();

  // preload x_0 into staging buffer 0 (branch-free)
  {
    float a0 = p0[0], a1 = p1[0], a2 = p2[0];
    xb[d0] = (_Float16)a0;
    xb[d1] = (_Float16)a1;
    xb[d2] = (_Float16)a2;
  }
  __syncthreads();

  v8f c0v = {}; v8f c2v = {};

  for (int t = 0; t < 512; ++t) {
    _Float16* xcur = xb + (t & 1) * 640;
    _Float16* xnxt = xb + ((t + 1) & 1) * 640;
    const int tn = (t < 511) ? (t + 1) : t;   // clamped prefetch index

    // A fragments for this step (x_t, h0_{t-1}, h1_{t-1})
    v16h xA   = ld_frag(xcur + m * 40 + hl * 16);
    v16h haA0 = ld_frag(hab + m * 72 + hl * 16);
    v16h haA1 = ld_frag(hab + m * 72 + 32 + hl * 16);
    v16h hbA0 = ld_frag(hbb + m * 72 + hl * 16);
    v16h hbA1 = ld_frag(hbb + m * 72 + 32 + hl * 16);

    // prefetch x_{t+1} from HBM; consumed (cvt+store) only at iteration end,
    // so the load latency hides behind the 28 WMMAs + cell updates below.
    float nx0 = p0[tn * 23];
    float nx1 = p1[tn * 23];
    float nx2 = p2[tn * 23];

    __syncthreads();   // all reads of hab/hbb done before this step's rewrites

    // ----- layer 0: gates(16x256) = x_t(16x32pad)@Wih0^T + h(16x64)@Whh0^T + b
    v8f acc[4];
#pragma unroll
    for (int g = 0; g < 4; ++g) {
      const int nt = g * 4 + wave;
      v8f a;
#pragma unroll
      for (int k = 0; k < 8; ++k) a[k] = bias0[g];
      a = wmma_f16(xA,   ld_frag(wb + nt * 512 + lane * 16), a);
      a = wmma_f16(haA0, ld_frag(wb + (16 + nt * 2 + 0) * 512 + lane * 16), a);
      a = wmma_f16(haA1, ld_frag(wb + (16 + nt * 2 + 1) * 512 + lane * 16), a);
      acc[g] = a;
    }
    // cell update L0, publish h1_t (f16) for both recurrence and layer-1 input
#pragma unroll
    for (int k = 0; k < 8; ++k) {
      float iv = fast_sig(acc[0][k]);
      float fv = fast_sig(acc[1][k]);
      float gv = fast_tanh(acc[2][k]);
      float ov = fast_sig(acc[3][k]);
      float cc = fv * c0v[k] + iv * gv;
      c0v[k] = cc;
      float hv = ov * fast_tanh(cc);
      hab[(k + 8 * hl) * 72 + wave * 16 + m] = (_Float16)hv;
    }
    __syncthreads();

    v16h h1A0 = ld_frag(hab + m * 72 + hl * 16);
    v16h h1A1 = ld_frag(hab + m * 72 + 32 + hl * 16);

    // ----- layer 1: gates = h1_t@Wih1^T + h2_{t-1}@Whh1^T + b
#pragma unroll
    for (int g = 0; g < 4; ++g) {
      const int nt = g * 4 + wave;
      v8f a;
#pragma unroll
      for (int k = 0; k < 8; ++k) a[k] = bias1[g];
      a = wmma_f16(h1A0, ld_frag(wb + (48 + nt * 2 + 0) * 512 + lane * 16), a);
      a = wmma_f16(h1A1, ld_frag(wb + (48 + nt * 2 + 1) * 512 + lane * 16), a);
      a = wmma_f16(hbA0, ld_frag(wb + (80 + nt * 2 + 0) * 512 + lane * 16), a);
      a = wmma_f16(hbA1, ld_frag(wb + (80 + nt * 2 + 1) * 512 + lane * 16), a);
      acc[g] = a;
    }
#pragma unroll
    for (int k = 0; k < 8; ++k) {
      float iv = fast_sig(acc[0][k]);
      float fv = fast_sig(acc[1][k]);
      float gv = fast_tanh(acc[2][k]);
      float ov = fast_sig(acc[3][k]);
      float cc = fv * c2v[k] + iv * gv;
      c2v[k] = cc;
      float hv = ov * fast_tanh(cc);
      hbb[(k + 8 * hl) * 72 + wave * 16 + m] = (_Float16)hv;
      if (t == 511) hfin[(k + 8 * hl) * 64 + wave * 16 + m] = hv;
    }

    // stage x_{t+1} into the other buffer (dead store on t==511, branch-free)
    xnxt[d0] = (_Float16)nx0;
    xnxt[d1] = (_Float16)nx1;
    xnxt[d2] = (_Float16)nx2;
    __syncthreads();
  }

  // ----- final FC: out[b] = h2_last . fcw + fcb -----
  if (tid < 16) {
    float s = fcb[0];
    const float* hf = hfin + tid * 64;
#pragma unroll 8
    for (int j = 0; j < 64; ++j) s += hf[j] * fcw[j];
    out[b0 + tid] = s;
  }
}

extern "C" void kernel_launch(void* const* d_in, const int* in_sizes, int n_in,
                              void* d_out, int out_size, void* d_ws, size_t ws_size,
                              hipStream_t stream) {
  (void)in_sizes; (void)n_in; (void)d_ws; (void)ws_size; (void)out_size;
  const float* x    = (const float*)d_in[0];
  const float* Wih0 = (const float*)d_in[1];
  const float* Whh0 = (const float*)d_in[2];
  const float* bih0 = (const float*)d_in[3];
  const float* bhh0 = (const float*)d_in[4];
  const float* Wih1 = (const float*)d_in[5];
  const float* Whh1 = (const float*)d_in[6];
  const float* bih1 = (const float*)d_in[7];
  const float* bhh1 = (const float*)d_in[8];
  const float* fcw  = (const float*)d_in[9];
  const float* fcb  = (const float*)d_in[10];
  float* outp = (float*)d_out;

  hipFuncSetAttribute((const void*)lstm2_fused,
                      hipFuncAttributeMaxDynamicSharedMemorySize, SMEM_BYTES);
  lstm2_fused<<<dim3(2048 / 16), dim3(128), SMEM_BYTES, stream>>>(
      x, Wih0, Whh0, bih0, bhh0, Wih1, Whh1, bih1, bhh1, fcw, fcb, outp);
}
